// VanillaMAPPOMicroCommResidualAdapterAgent_62457414419212
// MI455X (gfx1250) — matrix operand order repaired and checked by reading, not compiled, not run on hardware
//
#include <hip/hip_runtime.h>

// ---------------------------------------------------------------------------
// Types / fragment helpers
// ---------------------------------------------------------------------------
typedef __attribute__((ext_vector_type(16))) _Float16 v16h;
typedef __attribute__((ext_vector_type(8)))  float    v8f;

union HFrag { uint4 u[2]; v16h v; };

// A fragment (16x32 f16): lanes 0-15 row m hold K=0..7 (v0-3) & K=16..23 (v4-7);
// lanes 16-31 row m hold K=8..15 & K=24..31.
__device__ inline v16h ldA(const _Float16* rowp, int lane) {
  const int off = (lane & 16) ? 8 : 0;
  HFrag f;
  f.u[0] = *(const uint4*)(rowp + off);
  f.u[1] = *(const uint4*)(rowp + 16 + off);
  return f.v;
}
// B fragment (32x16 f16): lanes 0-15 col n hold K=0..7 (v0-3) & K=16..23 (v4-7);
// lanes 16-31 col n hold K=8..15 & K=24..31.
__device__ inline v16h ldB(const _Float16* rowp, int lane) {
  const int off = (lane & 16) ? 8 : 0;
  HFrag f;
  f.u[0] = *(const uint4*)(rowp + off);
  f.u[1] = *(const uint4*)(rowp + 16 + off);
  return f.v;
}

__device__ inline float sigf(float x) { return 1.f / (1.f + __expf(-x)); }

#define MROWS 131072           // BS * N = 8192 * 16
#define NEGV  (-1.0e10f)

// ---------------------------------------------------------------------------
// f32 -> f16 convert (grid-stride)
// ---------------------------------------------------------------------------
__global__ void cvt_f16(_Float16* __restrict__ dst, const float* __restrict__ src,
                        size_t n) {
  size_t i = (size_t)blockIdx.x * blockDim.x + threadIdx.x;
  size_t stride = (size_t)gridDim.x * blockDim.x;
  for (; i < n; i += stride) dst[i] = (_Float16)src[i];
}

// pack q/k/v weights into [192,256] (rows 160..191 zero) + bias [192]
__global__ void pack_qkv(_Float16* __restrict__ Wd, float* __restrict__ Bd,
                         const float* __restrict__ qw, const float* __restrict__ kw,
                         const float* __restrict__ vw, const float* __restrict__ qb,
                         const float* __restrict__ kb, const float* __restrict__ vb) {
  int idx = blockIdx.x * 256 + threadIdx.x;
  if (idx < 192 * 256) {
    int r = idx >> 8, c = idx & 255;
    float val = 0.f;
    if (r < 64)        val = qw[r * 256 + c];
    else if (r < 128)  val = kw[(r - 64) * 256 + c];
    else if (r < 160)  val = vw[(r - 128) * 256 + c];
    Wd[idx] = (_Float16)val;
  }
  if (idx < 192) {
    float b = 0.f;
    if (idx < 64)       b = qb[idx];
    else if (idx < 128) b = kb[idx - 64];
    else if (idx < 160) b = vb[idx - 128];
    Bd[idx] = b;
  }
}

// pad pol_w2 [20,256] -> [64,256], bias -> [64]
__global__ void pack_pol2(_Float16* __restrict__ Wd, float* __restrict__ Bd,
                          const float* __restrict__ w, const float* __restrict__ b) {
  int idx = blockIdx.x * 256 + threadIdx.x;
  if (idx < 64 * 256) {
    int r = idx >> 8, c = idx & 255;
    Wd[idx] = (_Float16)((r < 20) ? w[r * 256 + c] : 0.f);
  }
  if (idx < 64) Bd[idx] = (idx < 20) ? b[idx] : 0.f;
}

// ---------------------------------------------------------------------------
// Generic WMMA GEMM: C[M,N] = act(A[M,K](f16,lda) @ W[N,K]^T + bias)
// block = 256 thr (8 waves), tile 128x64, wave tile 32x32 (2x2 WMMA 16x16x32).
// Ping-pong LDS: prefetch next K-slice while WMMAs consume current one;
// one barrier per K-step.
// ACT: 0 none, 1 relu, 2 residual-fuse.  OUTF32/OUTF16 select store paths.
// ---------------------------------------------------------------------------
template <int ACT, bool OUTF32, bool OUTF16>
__global__ __launch_bounds__(256)
void gemm_f16(const _Float16* __restrict__ A, int lda,
              const _Float16* __restrict__ W,          // [N,K]
              const float* __restrict__ bias,
              float* __restrict__ Cf, _Float16* __restrict__ Ch,
              int ldc, int nstore, int K,
              const float* __restrict__ gate_vec,
              const float* __restrict__ add_src, float scale) {
  __shared__ _Float16 As[2][128 * 48];
  __shared__ _Float16 Bs[2][64 * 48];
  const int tid  = threadIdx.x;
  const int lane = tid & 31;
  const int w    = tid >> 5;
  const int wr   = w >> 1;   // 0..3
  const int wc   = w & 1;    // 0..1
  const int rowBase = blockIdx.y * 128;
  const int colBase = blockIdx.x * 64;

  const int ar = tid >> 1, aseg = (tid & 1) * 16;
  const int br = tid >> 2, bseg = (tid & 3) * 8;

  auto stage = [&](int buf, int k0) {
    const uint4* g = (const uint4*)(A + (size_t)(rowBase + ar) * lda + k0 + aseg);
    uint4* s = (uint4*)(As[buf] + ar * 48 + aseg);
    s[0] = g[0];
    s[1] = g[1];
    *(uint4*)(Bs[buf] + br * 48 + bseg) =
        *(const uint4*)(W + (size_t)(colBase + br) * K + k0 + bseg);
  };

  v8f acc[2][2] = {};
  stage(0, 0);
  __syncthreads();
  int buf = 0;
  const int mrow = lane & 15;
  for (int k0 = 0; k0 < K; k0 += 32) {
    if (k0 + 32 < K) stage(buf ^ 1, k0 + 32);   // prefetch next slice
    v16h af[2], bf[2];
#pragma unroll
    for (int tm = 0; tm < 2; ++tm)
      af[tm] = ldA(As[buf] + (wr * 32 + tm * 16 + mrow) * 48, lane);
#pragma unroll
    for (int tn = 0; tn < 2; ++tn)
      bf[tn] = ldB(Bs[buf] + (wc * 32 + tn * 16 + mrow) * 48, lane);
#pragma unroll
    for (int tm = 0; tm < 2; ++tm)
#pragma unroll
      for (int tn = 0; tn < 2; ++tn)
        acc[tm][tn] = __builtin_amdgcn_wmma_f32_16x16x32_f16(
            false, af[tm], false, bf[tn], (short)0, acc[tm][tn], false, false);
    __syncthreads();
    buf ^= 1;
  }
  // epilogue: C layout: lane 0-15 -> M=r, lane 16-31 -> M=r+8, N = lane&15
  const int halfAdd = (lane >> 4) * 8;
  const int ncol = lane & 15;
#pragma unroll
  for (int tm = 0; tm < 2; ++tm)
#pragma unroll
    for (int tn = 0; tn < 2; ++tn)
#pragma unroll
      for (int r = 0; r < 8; ++r) {
        int row = rowBase + wr * 32 + tm * 16 + r + halfAdd;
        int col = colBase + wc * 32 + tn * 16 + ncol;
        if (col >= nstore) continue;
        float v = acc[tm][tn][r] + bias[col];
        if (ACT == 1) v = fmaxf(v, 0.f);
        if (ACT == 2)
          v = add_src[(size_t)row * 256 + col] + v * scale * gate_vec[row];
        size_t o = (size_t)row * ldc + col;
        if (OUTF32) Cf[o] = v;
        if (OUTF16) Ch[o] = (_Float16)v;
      }
}

// ---------------------------------------------------------------------------
// Fused GRU: 6 simultaneous GEMM accumulations (wi/wh x r/z/n), tile 64x64,
// per wave 16(M)x32(N) per set.  Ping-pong LDS, one barrier per K-step.
// Writes h (f32 to d_out area, f16 to gate_in cols 0..255).
// ---------------------------------------------------------------------------
__global__ __launch_bounds__(256)
void gru_fused(const _Float16* __restrict__ X,    // [M,256] fc1 out
               const _Float16* __restrict__ Hin,  // [M,256]
               const float* __restrict__ Hin32,   // [M,256] original f32
               const _Float16* __restrict__ Wi,   // [768,256]
               const _Float16* __restrict__ Wh,   // [768,256]
               const float* __restrict__ bi, const float* __restrict__ bh,
               float* __restrict__ Hout32,        // ldc 256
               _Float16* __restrict__ GateIn) {   // [M,288] cols 0..255
  __shared__ _Float16 Xs[2][64 * 48];
  __shared__ _Float16 Hs[2][64 * 48];
  __shared__ _Float16 Ws[2][6][64 * 48];
  const int tid  = threadIdx.x;
  const int lane = tid & 31;
  const int w    = tid >> 5;
  const int wr   = w >> 1;   // 0..3 -> 16-row slice
  const int wc   = w & 1;    // 0..1 -> 32-col slice
  const int rowBase = blockIdx.y * 64;
  const int colBase = blockIdx.x * 64;   // within HID=256

  const int r   = tid >> 2;
  const int seg = (tid & 3) * 8;

  auto stage = [&](int buf, int k0) {
    *(uint4*)(Xs[buf] + r * 48 + seg) =
        *(const uint4*)(X + (size_t)(rowBase + r) * 256 + k0 + seg);
    *(uint4*)(Hs[buf] + r * 48 + seg) =
        *(const uint4*)(Hin + (size_t)(rowBase + r) * 256 + k0 + seg);
#pragma unroll
    for (int s = 0; s < 6; ++s) {
      const _Float16* Wsrc = (s < 3) ? Wi : Wh;
      int g = (s < 3) ? s : s - 3;
      *(uint4*)(Ws[buf][s] + r * 48 + seg) =
          *(const uint4*)(Wsrc + (size_t)(g * 256 + colBase + r) * 256 + k0 + seg);
    }
  };

  v8f acc[6][2] = {};
  stage(0, 0);
  __syncthreads();
  int buf = 0;
  const int mrow = lane & 15;
  for (int k0 = 0; k0 < 256; k0 += 32) {
    if (k0 + 32 < 256) stage(buf ^ 1, k0 + 32);
    v16h ax = ldA(Xs[buf] + (wr * 16 + mrow) * 48, lane);
    v16h ah = ldA(Hs[buf] + (wr * 16 + mrow) * 48, lane);
#pragma unroll
    for (int s = 0; s < 6; ++s) {
      v16h a = (s < 3) ? ax : ah;
#pragma unroll
      for (int tn = 0; tn < 2; ++tn) {
        v16h b = ldB(Ws[buf][s] + (wc * 32 + tn * 16 + mrow) * 48, lane);
        acc[s][tn] = __builtin_amdgcn_wmma_f32_16x16x32_f16(
            false, a, false, b, (short)0, acc[s][tn], false, false);
      }
    }
    __syncthreads();
    buf ^= 1;
  }
  const int halfAdd = (lane >> 4) * 8;
  const int ncol = lane & 15;
#pragma unroll
  for (int tn = 0; tn < 2; ++tn)
#pragma unroll
    for (int rr = 0; rr < 8; ++rr) {
      int row = rowBase + wr * 16 + rr + halfAdd;
      int col = colBase + wc * 32 + tn * 16 + ncol;
      float ir  = acc[0][tn][rr] + bi[col];
      float iz  = acc[1][tn][rr] + bi[col + 256];
      float in_ = acc[2][tn][rr] + bi[col + 512];
      float hr  = acc[3][tn][rr] + bh[col];
      float hz  = acc[4][tn][rr] + bh[col + 256];
      float hn  = acc[5][tn][rr] + bh[col + 512];
      float rg = sigf(ir + hr);
      float z  = sigf(iz + hz);
      float n  = tanhf(in_ + rg * hn);
      float hp = Hin32[(size_t)row * 256 + col];
      float hnew = (1.f - z) * n + z * hp;
      Hout32[(size_t)row * 256 + col] = hnew;
      GateIn[(size_t)row * 288 + col] = (_Float16)hnew;
    }
}

// ---------------------------------------------------------------------------
// Attention + top-8 + softmax + messages + LayerNorm.
// One block per batch (128 thr = 4 waves, wave = head).
// Writes normed f16 into gate_in cols 256..287.
// ---------------------------------------------------------------------------
__global__ __launch_bounds__(128)
void attn_kernel(const _Float16* __restrict__ QKV,  // [M,192]: q|k|v
                 const float* __restrict__ ln_g, const float* __restrict__ ln_b,
                 _Float16* __restrict__ GateIn) {
  __shared__ float alpha_s[4][16][17];
  __shared__ float v_s[4][16][8];
  __shared__ float flat_s[16][33];
  __shared__ float stats[16][2];
  const int tid = threadIdx.x;
  const int lane = tid & 31;
  const int h = tid >> 5;
  const int b = blockIdx.x;
  const size_t base = (size_t)b * 16 * 192;
  const int m = lane & 15;
  const int aoff = (lane & 16) ? 8 : 0;

  // Q as A fragment (K=16 padded to 32 with zeros), K-matrix as B fragment.
  HFrag qa, kb;
  qa.u[0] = *(const uint4*)(QKV + base + (size_t)m * 192 + h * 16 + aoff);
  qa.u[1] = make_uint4(0u, 0u, 0u, 0u);
  kb.u[0] = *(const uint4*)(QKV + base + (size_t)m * 192 + 64 + h * 16 + aoff);
  kb.u[1] = make_uint4(0u, 0u, 0u, 0u);
  v8f sc = {};
  sc = __builtin_amdgcn_wmma_f32_16x16x32_f16(false, qa.v, false, kb.v,
                                              (short)0, sc, false, false);
  // stage V (f32) in LDS
#pragma unroll
  for (int idx = lane; idx < 128; idx += 32) {
    int j = idx >> 3, vd = idx & 7;
    v_s[h][j][vd] = (float)QKV[base + (size_t)j * 192 + 128 + h * 8 + vd];
  }
  const int halfbase = lane & 16;
  const int j = lane & 15;
#pragma unroll
  for (int r = 0; r < 8; ++r) {
    int i = r + ((lane >> 4) << 3);
    float s = sc[r] * 0.25f;               // 1/sqrt(HD=16)
    if (i == j) s = NEGV;                  // mask self
    // top-8 over j via rank (stable tie-break by index, matches top_k)
    int cnt = 0;
#pragma unroll
    for (int t = 0; t < 16; ++t) {
      float o = __shfl(s, halfbase + t, 32);
      cnt += (o > s || (o == s && t < j)) ? 1 : 0;
    }
    if (cnt >= 8) s = NEGV;
    // softmax across the 16-lane half
    float mx = s;
#pragma unroll
    for (int d = 1; d < 16; d <<= 1) mx = fmaxf(mx, __shfl_xor(mx, d, 32));
    float e = __expf(s - mx);
    float sum = e;
#pragma unroll
    for (int d = 1; d < 16; d <<= 1) sum += __shfl_xor(sum, d, 32);
    alpha_s[h][i][j] = e / sum;
  }
  __syncthreads();
  // head messages -> flat [16][32]
#pragma unroll
  for (int e0 = tid; e0 < 512; e0 += 128) {
    int i = e0 >> 5, c = e0 & 31, hh = c >> 3, vd = c & 7;
    float a = 0.f;
#pragma unroll
    for (int jj = 0; jj < 16; ++jj) a += alpha_s[hh][i][jj] * v_s[hh][jj][vd];
    flat_s[i][c] = a;
  }
  __syncthreads();
  if (tid < 16) {
    float mu = 0.f;
#pragma unroll
    for (int c = 0; c < 32; ++c) mu += flat_s[tid][c];
    mu *= (1.f / 32.f);
    float var = 0.f;
#pragma unroll
    for (int c = 0; c < 32; ++c) { float d = flat_s[tid][c] - mu; var += d * d; }
    var *= (1.f / 32.f);
    stats[tid][0] = mu;
    stats[tid][1] = rsqrtf(var + 1e-5f);
  }
  __syncthreads();
#pragma unroll
  for (int e0 = tid; e0 < 512; e0 += 128) {
    int i = e0 >> 5, c = e0 & 31;
    float nv = (flat_s[i][c] - stats[i][0]) * stats[i][1] * ln_g[c] + ln_b[c];
    GateIn[((size_t)b * 16 + i) * 288 + 256 + c] = (_Float16)nv;
  }
}

// ---------------------------------------------------------------------------
// gate layer 2: sigmoid(gate_h . w2 + b2) per row, one wave per row
// ---------------------------------------------------------------------------
__global__ __launch_bounds__(256)
void gate2_kernel(const _Float16* __restrict__ GH, const float* __restrict__ w2,
                  const float* __restrict__ b2, float* __restrict__ gate) {
  int row = blockIdx.x * 8 + (threadIdx.x >> 5);
  int lane = threadIdx.x & 31;
  const _Float16* rp = GH + (size_t)row * 256;
  float acc = 0.f;
#pragma unroll
  for (int c = 0; c < 256; c += 32) acc += (float)rp[c + lane] * w2[c + lane];
#pragma unroll
  for (int d = 1; d < 32; d <<= 1) acc += __shfl_xor(acc, d, 32);
  if (lane == 0) gate[row] = sigf(acc + b2[0]);
}

// ---------------------------------------------------------------------------
// Workspace layout (bytes)
// ---------------------------------------------------------------------------
static constexpr size_t Msz = (size_t)MROWS;
static constexpr size_t OFF_WFC1   = 0;
static constexpr size_t OFF_WGI    = OFF_WFC1 + 256 * 128 * 2;
static constexpr size_t OFF_WGH    = OFF_WGI + 768 * 256 * 2;
static constexpr size_t OFF_WQKV   = OFF_WGH + 768 * 256 * 2;
static constexpr size_t OFF_BQKV   = OFF_WQKV + 192 * 256 * 2;
static constexpr size_t OFF_WG1    = OFF_BQKV + 1024;
static constexpr size_t OFF_WRES   = OFF_WG1 + 256 * 288 * 2;
static constexpr size_t OFF_WP1    = OFF_WRES + 256 * 32 * 2;
static constexpr size_t OFF_WP2    = OFF_WP1 + 256 * 256 * 2;
static constexpr size_t OFF_BP2    = OFF_WP2 + 64 * 256 * 2;
static constexpr size_t OFF_XIN    = OFF_BP2 + 512;
static constexpr size_t OFF_HIN    = OFF_XIN + Msz * 128 * 2;    // reused as p16
static constexpr size_t OFF_X      = OFF_HIN + Msz * 256 * 2;    // reused as gate_h
static constexpr size_t OFF_GATEIN = OFF_X + Msz * 256 * 2;
static constexpr size_t OFF_QKV    = OFF_GATEIN + Msz * 288 * 2;
static constexpr size_t OFF_GATEV  = OFF_QKV + Msz * 192 * 2;
static constexpr size_t OFF_FUSED  = OFF_GATEV + Msz * 4;

extern "C" void kernel_launch(void* const* d_in, const int* in_sizes, int n_in,
                              void* d_out, int out_size, void* d_ws, size_t ws_size,
                              hipStream_t stream) {
  (void)in_sizes; (void)n_in; (void)out_size; (void)ws_size;
  const float* inputs  = (const float*)d_in[0];
  const float* hidden  = (const float*)d_in[1];
  const float* fc1_w   = (const float*)d_in[3];
  const float* fc1_b   = (const float*)d_in[4];
  const float* gru_wi  = (const float*)d_in[5];
  const float* gru_wh  = (const float*)d_in[6];
  const float* gru_bi  = (const float*)d_in[7];
  const float* gru_bh  = (const float*)d_in[8];
  const float* q_w     = (const float*)d_in[9];
  const float* q_b     = (const float*)d_in[10];
  const float* k_w     = (const float*)d_in[11];
  const float* k_b     = (const float*)d_in[12];
  const float* v_w     = (const float*)d_in[13];
  const float* v_b     = (const float*)d_in[14];
  const float* ln_g    = (const float*)d_in[15];
  const float* ln_b    = (const float*)d_in[16];
  const float* gate_w1 = (const float*)d_in[17];
  const float* gate_b1 = (const float*)d_in[18];
  const float* gate_w2 = (const float*)d_in[19];
  const float* gate_b2 = (const float*)d_in[20];
  const float* res_w   = (const float*)d_in[21];
  const float* res_b   = (const float*)d_in[22];
  const float* pol_w1  = (const float*)d_in[23];
  const float* pol_b1  = (const float*)d_in[24];
  const float* pol_w2  = (const float*)d_in[25];
  const float* pol_b2  = (const float*)d_in[26];

  char* ws = (char*)d_ws;
  _Float16* wfc1   = (_Float16*)(ws + OFF_WFC1);
  _Float16* wgi    = (_Float16*)(ws + OFF_WGI);
  _Float16* wgh    = (_Float16*)(ws + OFF_WGH);
  _Float16* wqkv   = (_Float16*)(ws + OFF_WQKV);
  float*    bqkv   = (float*)   (ws + OFF_BQKV);
  _Float16* wg1    = (_Float16*)(ws + OFF_WG1);
  _Float16* wres   = (_Float16*)(ws + OFF_WRES);
  _Float16* wp1    = (_Float16*)(ws + OFF_WP1);
  _Float16* wp2    = (_Float16*)(ws + OFF_WP2);
  float*    bp2    = (float*)   (ws + OFF_BP2);
  _Float16* xin16  = (_Float16*)(ws + OFF_XIN);
  _Float16* hin16  = (_Float16*)(ws + OFF_HIN);
  _Float16* x16    = (_Float16*)(ws + OFF_X);
  _Float16* gatein = (_Float16*)(ws + OFF_GATEIN);
  _Float16* qkv16  = (_Float16*)(ws + OFF_QKV);
  float*    gatev  = (float*)   (ws + OFF_GATEV);
  _Float16* fused16= (_Float16*)(ws + OFF_FUSED);
  _Float16* gateh16= x16;     // reuse (x dead after GRU)
  _Float16* p16    = hin16;   // reuse (h_in f16 dead after GRU)

  float* logits = (float*)d_out;                       // [M,20]
  float* hout32 = (float*)d_out + Msz * 20;            // [M,256]

  // ---- precision conversion / packing ----
  cvt_f16<<<4096, 256, 0, stream>>>(xin16, inputs, Msz * 128);
  cvt_f16<<<4096, 256, 0, stream>>>(hin16, hidden, Msz * 256);
  cvt_f16<<<128, 256, 0, stream>>>(wfc1, fc1_w, 256 * 128);
  cvt_f16<<<768, 256, 0, stream>>>(wgi, gru_wi, 768 * 256);
  cvt_f16<<<768, 256, 0, stream>>>(wgh, gru_wh, 768 * 256);
  cvt_f16<<<288, 256, 0, stream>>>(wg1, gate_w1, 256 * 288);
  cvt_f16<<<32, 256, 0, stream>>>(wres, res_w, 256 * 32);
  cvt_f16<<<256, 256, 0, stream>>>(wp1, pol_w1, 256 * 256);
  pack_qkv<<<192, 256, 0, stream>>>(wqkv, bqkv, q_w, k_w, v_w, q_b, k_b, v_b);
  pack_pol2<<<64, 256, 0, stream>>>(wp2, bp2, pol_w2, pol_b2);

  // ---- fc1 + relu : [M,128] -> [M,256] f16 ----
  gemm_f16<1, false, true><<<dim3(4, MROWS / 128), 256, 0, stream>>>(
      xin16, 128, wfc1, fc1_b, nullptr, x16, 256, 256, 128,
      nullptr, nullptr, 0.f);

  // ---- fused GRU -> h (f32 into d_out, f16 into gate_in cols 0..255) ----
  gru_fused<<<dim3(4, MROWS / 64), 256, 0, stream>>>(
      x16, hin16, hidden, wgi, wgh, gru_bi, gru_bh, hout32, gatein);

  // ---- qkv projection: gate_in[:, :256] @ Wqkv^T -> [M,192] f16 ----
  gemm_f16<0, false, true><<<dim3(3, MROWS / 128), 256, 0, stream>>>(
      gatein, 288, wqkv, bqkv, nullptr, qkv16, 192, 192, 256,
      nullptr, nullptr, 0.f);

  // ---- attention + top8 + softmax + LN -> gate_in cols 256..287 ----
  attn_kernel<<<8192, 128, 0, stream>>>(qkv16, ln_g, ln_b, gatein);

  // ---- gate layer 1 (relu): [M,288] -> [M,256] f16 ----
  gemm_f16<1, false, true><<<dim3(4, MROWS / 128), 256, 0, stream>>>(
      gatein, 288, wg1, gate_b1, nullptr, gateh16, 256, 256, 288,
      nullptr, nullptr, 0.f);

  // ---- gate layer 2 (sigmoid) -> gate scalar per row ----
  gate2_kernel<<<MROWS / 8, 256, 0, stream>>>(gateh16, gate_w2, gate_b2, gatev);

  // ---- residual + fuse: fused = h + (normed@res_w^T + b)*scale*gate ----
  gemm_f16<2, false, true><<<dim3(4, MROWS / 128), 256, 0, stream>>>(
      gatein + 256, 288, wres, res_b, nullptr, fused16, 256, 256, 32,
      gatev, hout32, 0.1f);

  // ---- policy layer 1 (relu) ----
  gemm_f16<1, false, true><<<dim3(4, MROWS / 128), 256, 0, stream>>>(
      fused16, 256, wp1, pol_b1, nullptr, p16, 256, 256, 256,
      nullptr, nullptr, 0.f);

  // ---- policy layer 2 -> logits f32 [M,20] ----
  gemm_f16<0, true, false><<<dim3(1, MROWS / 128), 256, 0, stream>>>(
      p16, 256, wp2, bp2, logits, nullptr, 20, 20, 256,
      nullptr, nullptr, 0.f);
}
